// BiLSTM_CRF_78073915507288
// MI455X (gfx1250) — matrix operand order repaired and checked by reading, not compiled, not run on hardware
//
#include <hip/hip_runtime.h>
#include <hip/hip_bf16.h>
#include <stddef.h>
#include <stdint.h>

// ---------------------------------------------------------------------------
// BiLSTM-CRF for MI455X (gfx1250):
//  - batched projections via v_wmma_f32_16x16x32_bf16 (f32 accumulate)
//  - B operand staged in LDS via global_load_async_to_lds_b128 (ASYNCcnt),
//    double-buffered, 4x reuse across the 4 waves of each block
//  - sequential LSTM recurrence as persistent 1024-thread workgroup per dir
//  - single-wave Viterbi
// ---------------------------------------------------------------------------

#define L_SEQ  4096
#define E_DIM  300
#define E_PAD  320      // 300 -> mult of 32
#define H_DIM  512
#define H2_DIM 256
#define G_DIM  1024     // 4*H2
#define KV_IN  812      // E + H
#define KV_PAD 832      // -> mult of 32
#define T_TAG  20
#define T_PAD  64       // tag GEMM N padded to one 64-wide block
#define START_TAG 18
#define STOP_TAG  19

#define TILE_HALVES 2048    // 4 n-tiles * (32 lanes * 16 halves) per kt group

typedef __attribute__((ext_vector_type(8)))  __bf16 v8bf;
typedef __attribute__((ext_vector_type(16))) __bf16 v16bf;
typedef __attribute__((ext_vector_type(8)))  float  v8f;

__device__ __forceinline__ float sigm(float x) {
    return 1.0f / (1.0f + __expf(-x));
}

// ---------------------------------------------------------------------------
// Embedding gather: emb f32 (L x 300) and bf16 padded copy (L x 320)
// ---------------------------------------------------------------------------
__global__ void k_gather_embed(const int* __restrict__ sent,
                               const float* __restrict__ table,
                               float* __restrict__ emb,
                               __bf16* __restrict__ embbf) {
    const int l = blockIdx.x;
    const int j = threadIdx.x;          // 0..319
    const int s = sent[l];
    if (j < E_DIM) {
        float v = table[(size_t)s * E_DIM + j];
        emb[(size_t)l * E_DIM + j] = v;
        embbf[(size_t)l * E_PAD + j] = (__bf16)v;
    } else {
        embbf[(size_t)l * E_PAD + j] = (__bf16)0.0f;
    }
}

// ---------------------------------------------------------------------------
// Pack weight W (Nreal x Kreal, row-major f32) into per-lane WMMA B fragments:
// tile t = kt*(Npad/16)+nt holds 32 lanes x 16 bf16; lane<16 -> K = kt*32+j,
// lane>=16 -> K = kt*32+16+j, column n = nt*16 + (lane&15). Zero padded.
// ---------------------------------------------------------------------------
__global__ void k_pack_w(const float* __restrict__ W, __bf16* __restrict__ Bp,
                         int Kreal, int Kpad, int Nreal, int Npad) {
    const size_t total = (size_t)Kpad * Npad;
    size_t g = (size_t)blockIdx.x * blockDim.x + threadIdx.x;
    if (g >= total) return;
    const int j    = (int)(g & 15);
    const int lane = (int)((g >> 4) & 31);
    const int t    = (int)(g >> 9);
    const int ntn  = Npad >> 4;
    const int nt   = t % ntn;
    const int kt   = t / ntn;
    const int n = nt * 16 + (lane & 15);
    const int k = kt * 32 + ((lane >> 4) << 4) + j;
    float v = 0.0f;
    if (k < Kreal && n < Nreal) v = W[(size_t)n * Kreal + k];
    Bp[g] = (__bf16)v;
}

__global__ void k_add_bias(const float* __restrict__ a, const float* __restrict__ b,
                           float* __restrict__ o, int n) {
    int i = blockIdx.x * blockDim.x + threadIdx.x;
    if (i < n) o[i] = a[i] + b[i];
}

__global__ void k_pad_copy(const float* __restrict__ src, float* __restrict__ dst,
                           int n, int npad) {
    int i = blockIdx.x * blockDim.x + threadIdx.x;
    if (i < npad) dst[i] = (i < n) ? src[i] : 0.0f;
}

__global__ void k_zero_bf(__bf16* __restrict__ p, size_t n) {
    size_t i = (size_t)blockIdx.x * blockDim.x + threadIdx.x;
    if (i < n) p[i] = (__bf16)0.0f;
}

// copy f32 matrix (rows x cols, leading dim srcld) into bf16 matrix at column
// offset dstoff, leading dim dstld
__global__ void k_f32_to_bf(const float* __restrict__ src, int srcld, int cols,
                            __bf16* __restrict__ dst, int dstld, int dstoff) {
    const int r = blockIdx.x;
    for (int c = threadIdx.x; c < cols; c += blockDim.x)
        dst[(size_t)r * dstld + dstoff + c] = (__bf16)src[(size_t)r * srcld + c];
}

// ---------------------------------------------------------------------------
// WMMA GEMM: C(MxNpad f32) = A(MxKpad bf16 row-major) @ Bpack + bias.
// Block = 4 waves; each wave owns one 16-row tile; all waves share the same
// 64-column stripe. Per-kt B tile group (4KB) is async-DMA'd into LDS
// (double buffered) and consumed by all 4 waves. grid = (Npad/64, M/64).
// Requires M % 64 == 0 (all call sites satisfy this).
// ---------------------------------------------------------------------------
__global__ void __launch_bounds__(128)
k_gemm(const __bf16* __restrict__ A, const __bf16* __restrict__ Bp,
       const float* __restrict__ bias, float* __restrict__ C,
       int M, int Kpad, int Npad) {
    __shared__ __align__(16) __bf16 Bs[2][TILE_HALVES];

    const int tid  = threadIdx.x;
    const int lane = tid & 31;
    const int wave = tid >> 5;
    const int m0 = (blockIdx.y * 4 + wave) * 16;
    const int n0 = blockIdx.x * 64;
    const int ntn = Npad >> 4;
    const int nt0 = n0 >> 4;
    const int ktiles = Kpad >> 5;

    // async-DMA one kt group of B (4 n-tiles, 4KB) into LDS buffer `buf`
    auto prefetchB = [&](int kt, int buf) {
        const __bf16* __restrict__ g = Bp + ((size_t)kt * ntn + nt0) * 512;
        const uint32_t l0 = (uint32_t)(uintptr_t)(&Bs[buf][0]);
#pragma unroll
        for (int c = 0; c < 2; ++c) {
            const int off = (c * 128 + tid) * 8;               // halves (16B chunks)
            const uint32_t laddr = l0 + (uint32_t)off * 2;     // LDS byte address
            const uint64_t gaddr = (uint64_t)(uintptr_t)(g + off);
            asm volatile("global_load_async_to_lds_b128 %0, %1, off"
                         :: "v"(laddr), "v"(gaddr) : "memory");
        }
    };

    v8f acc[4] = {v8f{}, v8f{}, v8f{}, v8f{}};

    const int arow = m0 + (lane & 15);
    const int koff = (lane >> 4) << 4;               // 0 or 16 halves
    const __bf16* __restrict__ Abase = A + (size_t)arow * Kpad + koff;

    prefetchB(0, 0);
    for (int kt = 0; kt < ktiles; ++kt) {
        // publish buffer kt&1: my async writes done, then all waves' writes
        asm volatile("s_wait_asynccnt 0x0" ::: "memory");
        __syncthreads();
        // overlap next DMA with this iteration's WMMAs
        if (kt + 1 < ktiles) prefetchB(kt + 1, (kt + 1) & 1);
        if (kt + 2 < ktiles)
            __builtin_prefetch(Abase + (size_t)(kt + 2) * 32, 0, 1);

        v8bf alo = *(const v8bf*)(Abase + (size_t)kt * 32);
        v8bf ahi = *(const v8bf*)(Abase + (size_t)kt * 32 + 8);
        v16bf a = __builtin_shufflevector(alo, ahi,
            0,1,2,3,4,5,6,7,8,9,10,11,12,13,14,15);

        const __bf16* bs = &Bs[kt & 1][0];
#pragma unroll
        for (int q4 = 0; q4 < 4; ++q4) {
            const __bf16* bb = bs + ((q4 * 32 + lane) << 4);
            v8bf blo = *(const v8bf*)bb;
            v8bf bhi = *(const v8bf*)(bb + 8);
            v16bf b = __builtin_shufflevector(blo, bhi,
                0,1,2,3,4,5,6,7,8,9,10,11,12,13,14,15);
            acc[q4] = __builtin_amdgcn_wmma_f32_16x16x32_bf16(
                false, a, false, b, (short)0, acc[q4], false, false);
        }
        // next iteration's start barrier protects buffer reuse
    }

    const int mrow = m0 + ((lane >> 4) << 3);        // M = r + 8*(lane>=16)
#pragma unroll
    for (int q4 = 0; q4 < 4; ++q4) {
        const int ncol = n0 + q4 * 16 + (lane & 15);
        const float bv = bias ? bias[ncol] : 0.0f;
#pragma unroll
        for (int r = 0; r < 8; ++r)
            C[(size_t)(mrow + r) * Npad + ncol] = acc[q4][r] + bv;
    }
}

// ---------------------------------------------------------------------------
// Sequential LSTM: block 0 = forward, block 1 = backward. 1024 threads; thread
// t computes gate t (dot over h, 256). xproj already holds x@Wih^T + bih + bhh.
// Writes h into ht[p][0:256] (fwd) / ht[p][256:512] (bwd).
// ---------------------------------------------------------------------------
__global__ void __launch_bounds__(1024)
k_lstm(const float* __restrict__ xproj_f, const float* __restrict__ xproj_b,
       const float* __restrict__ Whh_f, const float* __restrict__ Whh_b,
       float* __restrict__ ht) {
    const int t = threadIdx.x;
    const bool back = (blockIdx.x != 0);
    const float* __restrict__ xp  = back ? xproj_b : xproj_f;
    const float* __restrict__ Whh = back ? Whh_b : Whh_f;

    __shared__ __align__(16) float h_lds[H2_DIM];
    __shared__ float g_lds[G_DIM];

    float c = 0.0f;
    if (t < H2_DIM) h_lds[t] = 0.0f;
    __syncthreads();

    const float4* __restrict__ wrow = (const float4*)(Whh + (size_t)t * H2_DIM);

    for (int step = 0; step < L_SEQ; ++step) {
        const int p = back ? (L_SEQ - 1 - step) : step;
        float g = xp[(size_t)p * G_DIM + t];
        const float4* h4 = (const float4*)h_lds;
#pragma unroll 8
        for (int k = 0; k < H2_DIM / 4; ++k) {
            float4 w = wrow[k];
            float4 h = h4[k];
            g = fmaf(w.x, h.x, g);
            g = fmaf(w.y, h.y, g);
            g = fmaf(w.z, h.z, g);
            g = fmaf(w.w, h.w, g);
        }
        g_lds[t] = g;
        __syncthreads();
        if (t < H2_DIM) {
            float ig = sigm(g_lds[t]);
            float fg = sigm(g_lds[H2_DIM + t]);
            float gg = tanhf(g_lds[2 * H2_DIM + t]);
            float og = sigm(g_lds[3 * H2_DIM + t]);
            c = fg * c + ig * gg;
            float h = og * tanhf(c);
            h_lds[t] = h;
            ht[(size_t)p * H_DIM + (back ? H2_DIM : 0) + t] = h;
        }
        __syncthreads();
    }
}

// ---------------------------------------------------------------------------
// Local attention (window of 6 neighbors). One block (256 thr) per position.
// Writes attn directly as bf16 into A_feat columns [512, 1024).
// ---------------------------------------------------------------------------
__global__ void k_attn(const float* __restrict__ q, const float* __restrict__ Km,
                       const float* __restrict__ Vm, const int* __restrict__ attn_id,
                       __bf16* __restrict__ A_feat) {
    const int l = blockIdx.x;
    const int tid = threadIdx.x;
    __shared__ float red[256];
    __shared__ float probs[8];
    __shared__ int idx_s[6];
    if (tid < 6) idx_s[tid] = attn_id[l * 6 + tid];
    __syncthreads();

    const float* __restrict__ qr = q + (size_t)l * H_DIM;
    const float inv_sqrt_h = 0.044194173824159216f;   // 1/sqrt(512)

    for (int w = 0; w < 6; ++w) {
        const float* __restrict__ Kr = Km + (size_t)idx_s[w] * H_DIM;
        red[tid] = fmaf(Kr[tid], qr[tid], Kr[tid + 256] * qr[tid + 256]);
        __syncthreads();
        for (int s = 128; s > 0; s >>= 1) {
            if (tid < s) red[tid] += red[tid + s];
            __syncthreads();
        }
        if (tid == 0) probs[w] = red[0] * inv_sqrt_h;
        __syncthreads();
    }
    if (tid == 0) {
        float m = probs[0];
        for (int w = 1; w < 6; ++w) m = fmaxf(m, probs[w]);
        float e[6], s = 0.0f;
        for (int w = 0; w < 6; ++w) { e[w] = __expf(probs[w] - m); s += e[w]; }
        float inv = 1.0f / s;
        for (int w = 0; w < 6; ++w) probs[w] = e[w] * inv;
    }
    __syncthreads();
    for (int h = tid; h < H_DIM; h += 256) {
        float a = 0.0f;
        for (int w = 0; w < 6; ++w)
            a = fmaf(probs[w], Vm[(size_t)idx_s[w] * H_DIM + h], a);
        A_feat[(size_t)l * G_DIM + H_DIM + h] = (__bf16)a;
    }
}

// ---------------------------------------------------------------------------
// Viterbi: single wave (20 tags). transitions row per lane in registers.
// ---------------------------------------------------------------------------
__global__ void k_viterbi(const float* __restrict__ feats, int ldf,
                          const float* __restrict__ trans,
                          int* __restrict__ bptrs, float* __restrict__ out) {
    const int t = threadIdx.x;
    __shared__ float fv[32];
    __shared__ float term[32];
    float tr[T_TAG];
    if (t < T_TAG)
        for (int j = 0; j < T_TAG; ++j) tr[j] = trans[t * T_TAG + j];
    fv[t] = -10000.0f;
    if (t == START_TAG) fv[t] = 0.0f;
    __syncthreads();

    for (int l = 0; l < L_SEQ; ++l) {
        float best = -3.4e38f, f2 = 0.0f;
        int arg = 0;
        if (t < T_TAG) {
            for (int j = 0; j < T_TAG; ++j) {
                float v = fv[j] + tr[j];
                if (v > best) { best = v; arg = j; }
            }
            f2 = best + feats[(size_t)l * ldf + t];
            bptrs[l * T_TAG + t] = arg;
        }
        __syncthreads();
        if (t < T_TAG) fv[t] = f2;
        __syncthreads();
    }

    if (t < T_TAG) term[t] = fv[t] + trans[STOP_TAG * T_TAG + t];
    __syncthreads();
    if (t == 0) {
        float best = term[0];
        int bl = 0;
        for (int j = 1; j < T_TAG; ++j)
            if (term[j] > best) { best = term[j]; bl = j; }
        out[0] = best;
        int tag = bl;
        for (int l = L_SEQ - 1; l >= 0; --l) {
            out[1 + l] = (float)tag;
            tag = bptrs[l * T_TAG + tag];
        }
    }
}

// ---------------------------------------------------------------------------
// Host orchestration
// ---------------------------------------------------------------------------
extern "C" void kernel_launch(void* const* d_in, const int* in_sizes, int n_in,
                              void* d_out, int out_size, void* d_ws, size_t ws_size,
                              hipStream_t stream) {
    const int*   sentence    = (const int*)  d_in[0];
    const float* embeddings  = (const float*)d_in[1];
    const float* Wih_f       = (const float*)d_in[2];
    const float* Whh_f       = (const float*)d_in[3];
    const float* bih_f       = (const float*)d_in[4];
    const float* bhh_f       = (const float*)d_in[5];
    const float* Wih_b       = (const float*)d_in[6];
    const float* Whh_b       = (const float*)d_in[7];
    const float* bih_b       = (const float*)d_in[8];
    const float* bhh_b       = (const float*)d_in[9];
    const float* Wq          = (const float*)d_in[10];
    const float* bq          = (const float*)d_in[11];
    const float* Wk          = (const float*)d_in[12];
    const float* bk          = (const float*)d_in[13];
    const float* Wv          = (const float*)d_in[14];
    const float* bv          = (const float*)d_in[15];
    const float* Wt          = (const float*)d_in[16];
    const float* bt          = (const float*)d_in[17];
    const float* transitions = (const float*)d_in[18];
    const int*   attn_id     = (const int*)  d_in[19];
    float* out = (float*)d_out;

    // bump allocator on d_ws (256B aligned)
    char* wsp = (char*)d_ws;
    auto alloc = [&](size_t bytes) -> void* {
        void* p = (void*)wsp;
        wsp += (bytes + 255) & ~(size_t)255;
        return p;
    };

    float*  emb      = (float*) alloc((size_t)L_SEQ * E_DIM * 4);
    __bf16* embbf    = (__bf16*)alloc((size_t)L_SEQ * E_PAD * 2);
    __bf16* WihF_p   = (__bf16*)alloc((size_t)E_PAD * G_DIM * 2);
    __bf16* WihB_p   = (__bf16*)alloc((size_t)E_PAD * G_DIM * 2);
    float*  xproj_f  = (float*) alloc((size_t)L_SEQ * G_DIM * 4);
    float*  xproj_b  = (float*) alloc((size_t)L_SEQ * G_DIM * 4);
    float*  ht       = (float*) alloc((size_t)L_SEQ * H_DIM * 4);
    __bf16* A_q      = (__bf16*)alloc((size_t)L_SEQ * H_DIM * 2);
    __bf16* A_kv     = (__bf16*)alloc((size_t)L_SEQ * KV_PAD * 2);
    __bf16* Wq_p     = (__bf16*)alloc((size_t)H_DIM * H_DIM * 2);
    __bf16* Wk_p     = (__bf16*)alloc((size_t)KV_PAD * H_DIM * 2);
    __bf16* Wv_p     = (__bf16*)alloc((size_t)KV_PAD * H_DIM * 2);
    __bf16* Wt_p     = (__bf16*)alloc((size_t)G_DIM * T_PAD * 2);
    float*  qbuf     = (float*) alloc((size_t)L_SEQ * H_DIM * 4);
    float*  Kbuf     = (float*) alloc((size_t)L_SEQ * H_DIM * 4);
    float*  Vbuf     = (float*) alloc((size_t)L_SEQ * H_DIM * 4);
    __bf16* A_feat   = (__bf16*)alloc((size_t)L_SEQ * G_DIM * 2);
    float*  feats    = (float*) alloc((size_t)L_SEQ * T_PAD * 4);
    int*    bptrs    = (int*)   alloc((size_t)L_SEQ * T_TAG * 4);
    float*  bcomb_f  = (float*) alloc(G_DIM * 4);
    float*  bcomb_b  = (float*) alloc(G_DIM * 4);
    float*  bt_pad   = (float*) alloc(T_PAD * 4);

    // 1) embedding gather (+ bf16 padded copy)
    k_gather_embed<<<L_SEQ, E_PAD, 0, stream>>>(sentence, embeddings, emb, embbf);

    // 2) combined biases, padded tag bias
    k_add_bias<<<(G_DIM + 255) / 256, 256, 0, stream>>>(bih_f, bhh_f, bcomb_f, G_DIM);
    k_add_bias<<<(G_DIM + 255) / 256, 256, 0, stream>>>(bih_b, bhh_b, bcomb_b, G_DIM);
    k_pad_copy<<<1, T_PAD, 0, stream>>>(bt, bt_pad, T_TAG, T_PAD);

    // 3) pack all weights into WMMA B-fragment layout (bf16)
    auto pack = [&](const float* W, __bf16* Bp, int Kr, int Kp, int Nr, int Np) {
        size_t total = (size_t)Kp * Np;
        k_pack_w<<<(unsigned)((total + 255) / 256), 256, 0, stream>>>(W, Bp, Kr, Kp, Nr, Np);
    };
    pack(Wih_f, WihF_p, E_DIM, E_PAD, G_DIM, G_DIM);
    pack(Wih_b, WihB_p, E_DIM, E_PAD, G_DIM, G_DIM);
    pack(Wq,    Wq_p,   H_DIM, H_DIM, H_DIM, H_DIM);
    pack(Wk,    Wk_p,   KV_IN, KV_PAD, H_DIM, H_DIM);
    pack(Wv,    Wv_p,   KV_IN, KV_PAD, H_DIM, H_DIM);
    pack(Wt,    Wt_p,   G_DIM, G_DIM, T_TAG, T_PAD);

    // 4) x-projections for both LSTM directions (WMMA, bias fused)
    auto gemm = [&](const __bf16* A, const __bf16* Bp, const float* bias,
                    float* C, int M, int Kp, int Np) {
        dim3 grid(Np / 64, M / 64);
        k_gemm<<<grid, 128, 0, stream>>>(A, Bp, bias, C, M, Kp, Np);
    };
    gemm(embbf, WihF_p, bcomb_f, xproj_f, L_SEQ, E_PAD, G_DIM);
    gemm(embbf, WihB_p, bcomb_b, xproj_b, L_SEQ, E_PAD, G_DIM);

    // 5) sequential bidirectional LSTM (fwd block 0, bwd block 1)
    k_lstm<<<2, 1024, 0, stream>>>(xproj_f, xproj_b, Whh_f, Whh_b, ht);

    // 6) build bf16 activation matrices
    k_zero_bf<<<(unsigned)(((size_t)L_SEQ * KV_PAD + 255) / 256), 256, 0, stream>>>(
        A_kv, (size_t)L_SEQ * KV_PAD);
    k_f32_to_bf<<<L_SEQ, 256, 0, stream>>>(emb, E_DIM, E_DIM, A_kv, KV_PAD, 0);
    k_f32_to_bf<<<L_SEQ, 256, 0, stream>>>(ht, H_DIM, H_DIM, A_kv, KV_PAD, E_DIM);
    k_f32_to_bf<<<L_SEQ, 256, 0, stream>>>(ht, H_DIM, H_DIM, A_q, H_DIM, 0);
    k_f32_to_bf<<<L_SEQ, 256, 0, stream>>>(ht, H_DIM, H_DIM, A_feat, G_DIM, 0);

    // 7) Q/K/V projections (WMMA)
    gemm(A_q,  Wq_p, bq, qbuf, L_SEQ, H_DIM, H_DIM);
    gemm(A_kv, Wk_p, bk, Kbuf, L_SEQ, KV_PAD, H_DIM);
    gemm(A_kv, Wv_p, bv, Vbuf, L_SEQ, KV_PAD, H_DIM);

    // 8) local attention -> bf16 columns [512,1024) of A_feat
    k_attn<<<L_SEQ, 256, 0, stream>>>(qbuf, Kbuf, Vbuf, attn_id, A_feat);

    // 9) tag features (WMMA, N padded 20 -> 64)
    gemm(A_feat, Wt_p, bt_pad, feats, L_SEQ, G_DIM, T_PAD);

    // 10) Viterbi decode -> out[0] = score, out[1..4096] = path
    k_viterbi<<<1, 32, 0, stream>>>(feats, T_PAD, transitions, bptrs, out);
}
